// HeadAttention_738734374917
// MI455X (gfx1250) — matrix operand order repaired and checked by reading, not compiled, not run on hardware
//
#include <hip/hip_runtime.h>
#include <hip/hip_fp16.h>

typedef __attribute__((ext_vector_type(16))) _Float16 v16h;
typedef __attribute__((ext_vector_type(8)))  _Float16 v8h;
typedef __attribute__((ext_vector_type(8)))  float    v8f;

static constexpr int Bb  = 8;
static constexpr int Nn  = 2048;
static constexpr int Dd  = 1024;
static constexpr int DHd = 64;
static constexpr float SCALE = 0.125f; // 1/sqrt(64)

// ---- WMMA operand loaders (per ISA 7.12.2 VGPR layouts, wave32) ------------

// A operand: 16x32 f16 tile, source row-major with leading dim lda (elements).
// lane<16: rows M=lane, K {0..7} and {16..23}; lane>=16: rows M=lane-16,
// K {8..15} and {24..31}.  Both halves are 16B-contiguous runs.
__device__ __forceinline__ v16h load_a16(const _Float16* base, int lda, int lane) {
    const int row = lane & 15;
    const int kb  = (lane < 16) ? 0 : 8;
    const _Float16* p = base + (size_t)row * lda + kb;
    v8h lo = *(const v8h*)(p);
    v8h hi = *(const v8h*)(p + 16);
    v16h a;
#pragma unroll
    for (int t = 0; t < 8; ++t) { a[t] = lo[t]; a[8 + t] = hi[t]; }
    return a;
}

// B operand: 32x16 f16 tile. Source must be K-major ("Bt"): Bt[n][k], leading
// dim ldb. lane<16: col n=lane, K {0..7} and {16..23}; lane>=16: col n=lane-16, K {8..15} and {24..31}.
__device__ __forceinline__ v16h load_b16(const _Float16* baseT, int ldb, int lane) {
    const int n  = lane & 15;
    const int kb = (lane < 16) ? 0 : 8;
    const _Float16* p = baseT + (size_t)n * ldb + kb;
    v8h lo = *(const v8h*)(p);
    v8h hi = *(const v8h*)(p + 16);
    v16h b;
#pragma unroll
    for (int t = 0; t < 8; ++t) { b[t] = lo[t]; b[8 + t] = hi[t]; }
    return b;
}

__device__ __forceinline__ v8f wmma_f16(v16h a, v16h b, v8f c) {
    return __builtin_amdgcn_wmma_f32_16x16x32_f16(false, a, false, b,
                                                  (short)0, c, false, false);
}

// ---- Pass 0: dtype conversion ----------------------------------------------

__global__ void ha_cvt_x(const float* __restrict__ x, _Float16* __restrict__ xh, int n) {
    int i = blockIdx.x * blockDim.x + threadIdx.x;
    if (i < n) xh[i] = (_Float16)x[i];
}

// WtAll[mat][n][k] = W_mat[k][n]  (K-major f16, ready as WMMA B operand)
__global__ void ha_build_wt(const float* __restrict__ Wq, const float* __restrict__ Wk,
                            const float* __restrict__ Wv, _Float16* __restrict__ WtAll) {
    int i = blockIdx.x * blockDim.x + threadIdx.x;
    if (i >= 3 * DHd * Dd) return;
    int mat = i / (DHd * Dd);
    int r   = i % (DHd * Dd);
    int n   = r / Dd;
    int k   = r % Dd;
    const float* W = (mat == 0) ? Wq : (mat == 1) ? Wk : Wv;
    WtAll[i] = (_Float16)W[(size_t)k * DHd + n];
}

// ---- Pass 1: QKV projection GEMM (M=B*N, K=D, N=DH), bias fused ------------
// 128 threads = 4 independent waves; each wave owns one 16x16 output tile.

__global__ void ha_proj(const _Float16* __restrict__ xh, const _Float16* __restrict__ WtAll,
                        const float* __restrict__ bq, const float* __restrict__ bk,
                        const float* __restrict__ bv,
                        _Float16* __restrict__ Qh, _Float16* __restrict__ Kh,
                        _Float16* __restrict__ Vh) {
    const int lane = threadIdx.x & 31;
    const int wid  = threadIdx.x >> 5;
    const int tidx = blockIdx.x * 4 + wid;          // (BN/16)*(DH/16) tiles
    const int tN   = (tidx & 3) * 16;               // DH/16 == 4
    const int tM   = (tidx >> 2) * 16;
    const int mat  = blockIdx.y;

    const _Float16* Wt  = WtAll + (size_t)mat * DHd * Dd;
    const float* bias   = (mat == 0) ? bq : (mat == 1) ? bk : bv;
    _Float16* outp      = (mat == 0) ? Qh : (mat == 1) ? Kh : Vh;

    v8f c = {};
#pragma unroll 4
    for (int k0 = 0; k0 < Dd; k0 += 32) {
        v16h a = load_a16(xh + (size_t)tM * Dd + k0, Dd, lane);
        v16h b = load_b16(Wt + (size_t)tN * Dd + k0, Dd, lane);
        c = wmma_f16(a, b, c);
    }
    const float bv_ = bias[tN + (lane & 15)];
    const int n    = lane & 15;
    const int mof  = (lane < 16) ? 0 : 8;
#pragma unroll
    for (int r = 0; r < 8; ++r)
        outp[(size_t)(tM + mof + r) * DHd + tN + n] = (_Float16)(c[r] + bv_);
}

// ---- Pass 2: column stats m_j, Z_j over the query axis i; build Vt/Z -------
// One wave per 16-column j-tile; the K-tile B-fragments are loop-invariant and
// held in registers while Q streams through.  Online softmax reduction in the
// C-layout; lane-halves merged with shfl_xor(16).

__global__ void ha_colstats(const _Float16* __restrict__ Qh, const _Float16* __restrict__ Kh,
                            const _Float16* __restrict__ Vh,
                            float* __restrict__ mArr, float* __restrict__ ZArr,
                            _Float16* __restrict__ Vth) {
    const int lane = threadIdx.x & 31;
    const int wid  = threadIdx.x >> 5;
    const int tidx = blockIdx.x * 4 + wid;          // B*(N/16) tiles
    const int b    = tidx >> 7;                      // N/16 == 128
    const int jt   = tidx & 127;

    const _Float16* Qb = Qh + (size_t)b * Nn * DHd;
    const _Float16* Kb = Kh + (size_t)(b * Nn + jt * 16) * DHd;

    // K fragments for these 16 columns: invariant over the whole i-loop.
    const v16h kb0 = load_b16(Kb + 0,  DHd, lane);
    const v16h kb1 = load_b16(Kb + 32, DHd, lane);

    float mrun = -3.0e38f, Zrun = 0.0f;
    for (int it = 0; it < Nn / 16; ++it) {
        v16h a0 = load_a16(Qb + (size_t)(it * 16) * DHd + 0,  DHd, lane);
        v16h a1 = load_a16(Qb + (size_t)(it * 16) * DHd + 32, DHd, lane);
        v8f c = {};
        c = wmma_f16(a0, kb0, c);
        c = wmma_f16(a1, kb1, c);

        float tmax = mrun;
#pragma unroll
        for (int r = 0; r < 8; ++r) tmax = fmaxf(tmax, c[r] * SCALE);
        float s = 0.0f;
#pragma unroll
        for (int r = 0; r < 8; ++r) s += __expf(c[r] * SCALE - tmax);
        Zrun = Zrun * __expf(mrun - tmax) + s;
        mrun = tmax;
    }
    // merge the two lane-halves (rows 0-7 vs 8-15 of each tile column)
    float mo = __shfl_xor(mrun, 16, 32);
    float Zo = __shfl_xor(Zrun, 16, 32);
    float mm = fmaxf(mrun, mo);
    float ZZ = Zrun * __expf(mrun - mm) + Zo * __expf(mo - mm);

    const int j = jt * 16 + (lane & 15);
    if (lane < 16) { mArr[b * Nn + j] = mm; ZArr[b * Nn + j] = ZZ; }

    // Vt[b][d][j] = V[b][j][d] / Z_j  (K-major so it loads as a B operand)
    const float zinv = 1.0f / ZZ;
    const int dlo = (lane < 16) ? 0 : 32;
#pragma unroll 8
    for (int d = dlo; d < dlo + 32; ++d)
        Vth[((size_t)b * DHd + d) * Nn + j] =
            (_Float16)((float)Vh[((size_t)(b * Nn + j)) * DHd + d] * zinv);
}

// ---- Pass 3: out = exp(S - m_j) @ Vt ---------------------------------------
// One wave per 16-row query tile; Q fragments held in registers across the
// whole j-loop.  Recompute S in 16x32 chunks (cheaper than a 192 MB S/P HBM
// round-trip), stage P through LDS (C-layout -> A-layout), then WMMA-
// accumulate into a 16x64 f32 tile.

__global__ void ha_attn_out(const _Float16* __restrict__ Qh, const _Float16* __restrict__ Kh,
                            const _Float16* __restrict__ Vth, const float* __restrict__ mArr,
                            float* __restrict__ out) {
    __shared__ _Float16 plds[4 * 16 * 32];           // one 16x32 P tile per wave
    const int lane = threadIdx.x & 31;
    const int wid  = threadIdx.x >> 5;
    const int tidx = blockIdx.x * 4 + wid;           // B*(N/16) tiles
    const int b    = tidx >> 7;
    const int it   = tidx & 127;
    _Float16* P    = plds + wid * (16 * 32);

    const _Float16* Qt = Qh + (size_t)(b * Nn + it * 16) * DHd;
    const _Float16* Kb = Kh + (size_t)b * Nn * DHd;
    const _Float16* Vb = Vth + (size_t)b * DHd * Nn;
    const float*    mb = mArr + b * Nn;

    const int n   = lane & 15;
    const int mof = (lane < 16) ? 0 : 8;

    // Q fragments: invariant over the whole j-loop.
    const v16h qa0 = load_a16(Qt + 0,  DHd, lane);
    const v16h qa1 = load_a16(Qt + 32, DHd, lane);

    v8f acc0 = {}, acc1 = {}, acc2 = {}, acc3 = {};

    for (int jc = 0; jc < Nn; jc += 32) {
        // ---- S tile (16 x 32) in two 16x16 WMMA sub-tiles, exp into LDS ----
#pragma unroll
        for (int jn = 0; jn < 32; jn += 16) {
            v8f s = {};
            s = wmma_f16(qa0, load_b16(Kb + (size_t)(jc + jn) * DHd + 0,  DHd, lane), s);
            s = wmma_f16(qa1, load_b16(Kb + (size_t)(jc + jn) * DHd + 32, DHd, lane), s);
            const float mj = mb[jc + jn + n];
#pragma unroll
            for (int r = 0; r < 8; ++r) {
                float p = __expf(s[r] * SCALE - mj);
                P[(mof + r) * 32 + jn + n] = (_Float16)p;
            }
        }
        asm volatile("s_wait_dscnt 0" ::: "memory");  // same-wave LDS RAW fence

        // ---- reload P as an A operand (16x32, lda=32) ----------------------
        v16h ap;
        {
            const int row = lane & 15;
            const int kb  = (lane < 16) ? 0 : 8;
#pragma unroll
            for (int t = 0; t < 8; ++t) {
                ap[t]     = P[row * 32 + kb + t];
                ap[8 + t] = P[row * 32 + kb + 16 + t];
            }
        }
        // ---- accumulate P @ Vt over this 32-wide j chunk (N = DH = 64) -----
        acc0 = wmma_f16(ap, load_b16(Vb + (size_t)(0 * 16) * Nn + jc, Nn, lane), acc0);
        acc1 = wmma_f16(ap, load_b16(Vb + (size_t)(1 * 16) * Nn + jc, Nn, lane), acc1);
        acc2 = wmma_f16(ap, load_b16(Vb + (size_t)(2 * 16) * Nn + jc, Nn, lane), acc2);
        acc3 = wmma_f16(ap, load_b16(Vb + (size_t)(3 * 16) * Nn + jc, Nn, lane), acc3);
        asm volatile("" ::: "memory");                // keep LDS reuse ordered
    }

    float* ob = out + (size_t)(b * Nn + it * 16) * DHd;
#pragma unroll
    for (int r = 0; r < 8; ++r) {
        ob[(size_t)(mof + r) * DHd + 0 * 16 + n] = acc0[r];
        ob[(size_t)(mof + r) * DHd + 1 * 16 + n] = acc1[r];
        ob[(size_t)(mof + r) * DHd + 2 * 16 + n] = acc2[r];
        ob[(size_t)(mof + r) * DHd + 3 * 16 + n] = acc3[r];
    }
}

// ---- Host launcher ----------------------------------------------------------

extern "C" void kernel_launch(void* const* d_in, const int* in_sizes, int n_in,
                              void* d_out, int out_size, void* d_ws, size_t ws_size,
                              hipStream_t stream) {
    const float* x  = (const float*)d_in[0];
    const float* Wq = (const float*)d_in[1];
    const float* bq = (const float*)d_in[2];
    const float* Wk = (const float*)d_in[3];
    const float* bk = (const float*)d_in[4];
    const float* Wv = (const float*)d_in[5];
    const float* bv = (const float*)d_in[6];
    float* out = (float*)d_out;

    size_t off = 0;
    auto carve = [&](size_t bytes) {
        void* p = (char*)d_ws + off;
        off += (bytes + 255) & ~(size_t)255;
        return p;
    };
    _Float16* xh    = (_Float16*)carve((size_t)Bb * Nn * Dd * 2);   // 32 MB
    _Float16* WtAll = (_Float16*)carve((size_t)3 * DHd * Dd * 2);   // 384 KB
    _Float16* Qh    = (_Float16*)carve((size_t)Bb * Nn * DHd * 2);  // 2 MB
    _Float16* Kh    = (_Float16*)carve((size_t)Bb * Nn * DHd * 2);  // 2 MB
    _Float16* Vh    = (_Float16*)carve((size_t)Bb * Nn * DHd * 2);  // 2 MB
    _Float16* Vth   = (_Float16*)carve((size_t)Bb * DHd * Nn * 2);  // 2 MB
    float*    mArr  = (float*)carve((size_t)Bb * Nn * 4);
    float*    ZArr  = (float*)carve((size_t)Bb * Nn * 4);
    (void)ws_size; (void)in_sizes; (void)n_in; (void)out_size;

    // Pass 0: conversions
    {
        int n = Bb * Nn * Dd;
        ha_cvt_x<<<(n + 255) / 256, 256, 0, stream>>>(x, xh, n);
        int m = 3 * DHd * Dd;
        ha_build_wt<<<(m + 255) / 256, 256, 0, stream>>>(Wq, Wk, Wv, WtAll);
    }
    // Pass 1: QKV projections.  tiles = (BN/16)*(DH/16) = 4096, 4 waves/block
    ha_proj<<<dim3(1024, 3), 128, 0, stream>>>(xh, WtAll, bq, bk, bv, Qh, Kh, Vh);
    // Pass 2: column softmax stats + Vt/Z.  tiles = B*N/16 = 1024
    ha_colstats<<<256, 128, 0, stream>>>(Qh, Kh, Vh, mArr, ZArr, Vth);
    // Pass 3: output.  tiles = B*N/16 = 1024
    ha_attn_out<<<256, 128, 0, stream>>>(Qh, Kh, Vth, mArr, out);
}
